// SG_87101936763081
// MI455X (gfx1250) — matrix operand order repaired and checked by reading, not compiled, not run on hardware
//
#include <hip/hip_runtime.h>
#include <cmath>

// ---------------- problem constants ----------------
#define Bb    8
#define Nn    1024
#define Kk    16
#define CINc  128
#define DIMd  64
#define HIDh  256
#define NK    (Nn * Kk)      // 16384
#define M2    (Nn * 2)       // 2048
#define FEATf 512

typedef __attribute__((ext_vector_type(16))) _Float16 v16h;
typedef __attribute__((ext_vector_type(8)))  float    v8f;

// =====================================================================
// Generic WMMA GEMM: Out[b][o][m] = epilogue( sum_c W[o,c] * X[b][c][m] )
//   W addressed as W + o*wRow + c*wCol  (handles transposed / strided weights)
//   epilogue: (+bias[b?,o]) -> (*scale[o]*scaleMul + shift[o]) -> relu
//             -> (+ res[b][o][m/resDiv])
// Each wave32 computes a 16x64 output strip: A fragment loaded once per
// K-step (float4-vectorized when wCol==1), reused by 4 WMMAs.
// All strides are 32-bit so the backend can use scalar-base + 32-bit
// voffset addressing (no 64-bit VGPR address pairs, no acc spills).
// =====================================================================
__global__ void wmma_gemm_kernel(
    const float* __restrict__ W, int wRow, int wCol,
    const float* __restrict__ X, int ldx, int xB,
    float* __restrict__ Out, int ldo, int oB,
    const float* __restrict__ bias, int biasB,
    const float* __restrict__ scale, const float* __restrict__ shift, float scaleMul,
    const float* __restrict__ res, int ldr, int rB, int resDiv,
    int Cout, int Cin, int M, int relu)
{
    const int b      = blockIdx.y;
    const int wave   = threadIdx.x >> 5;
    const int lane   = threadIdx.x & 31;
    const int strips = M >> 6;                 // 64-wide column strips
    const int tilesO = Cout >> 4;
    const int t = blockIdx.x * (blockDim.x >> 5) + wave;
    if (t >= strips * tilesO) return;          // wave-uniform: EXEC stays full
    const int to = t / strips;
    const int ts = t - to * strips;

    const float* __restrict__ Xb = X + b * xB;
    float*       __restrict__ Ob = Out + b * oB;

    const int r    = lane & 15;
    const int hi   = lane >> 4;
    const int aRow = to * 16 + r;              // W row this lane loads (A fragment)
    const int col0 = ts * 64 + r;              // first X column this lane loads

    v8f acc[4] = {};
    const bool densW = (wCol == 1);

    for (int k0 = 0; k0 < Cin; k0 += 32) {
        // ---- A fragment: 16-bit 16x32 layout.
        // lane<16: K 0-7 & 16-23 ; lane>=16: K 8-15 & 24-31
        v16h af;
        if (densW) {
            const float* ap = W + aRow * wRow + k0 + hi * 8;
            float4 a0 = *(const float4*)(ap);
            float4 a1 = *(const float4*)(ap + 4);
            float4 a2 = *(const float4*)(ap + 16);
            float4 a3 = *(const float4*)(ap + 20);
            af[0]  = (_Float16)a0.x; af[1]  = (_Float16)a0.y;
            af[2]  = (_Float16)a0.z; af[3]  = (_Float16)a0.w;
            af[4]  = (_Float16)a1.x; af[5]  = (_Float16)a1.y;
            af[6]  = (_Float16)a1.z; af[7]  = (_Float16)a1.w;
            af[8]  = (_Float16)a2.x; af[9]  = (_Float16)a2.y;
            af[10] = (_Float16)a2.z; af[11] = (_Float16)a2.w;
            af[12] = (_Float16)a3.x; af[13] = (_Float16)a3.y;
            af[14] = (_Float16)a3.z; af[15] = (_Float16)a3.w;
        } else {
            const int wbase = aRow * wRow + k0 * wCol;
#pragma unroll
            for (int j = 0; j < 16; ++j) {
                int kk = hi * 8 + j + ((j >= 8) ? 8 : 0);
                af[j] = (_Float16)W[wbase + kk * wCol];
            }
        }
        // ---- B fragments: lanes 0-15 hold K 0-7 & 16-23 of column, lanes 16-31 K 8-15 & 24-31
        const float* __restrict__ xp = Xb + (k0 + hi * 8) * ldx + col0;
#pragma unroll
        for (int u = 0; u < 4; ++u) {
            float xv[16];
#pragma unroll
            for (int j = 0; j < 16; ++j)
                xv[j] = xp[(j + ((j >= 8) ? 8 : 0)) * ldx + u * 16];
            v16h bfv;
#pragma unroll
            for (int j = 0; j < 16; ++j)
                bfv[j] = (_Float16)xv[j];
            acc[u] = __builtin_amdgcn_wmma_f32_16x16x32_f16(false, af, false, bfv,
                                                            (short)0, acc[u], false, false);
        }
    }

    // ---- epilogue. C/D layout: lane<16 -> col=lane, row=v ; lane>=16 -> row=v+8
#pragma unroll
    for (int v = 0; v < 8; ++v) {
        const int oRow = to * 16 + v + hi * 8;
        const float bi = bias  ? bias[b * biasB + oRow] : 0.0f;
        const float sc = scale ? scale[oRow] * scaleMul : 1.0f;
        const float sh = scale ? shift[oRow] : 0.0f;
        float* __restrict__ orow = Ob + oRow * ldo;
        const float* __restrict__ rrow = res ? (res + b * rB + oRow * ldr) : nullptr;
#pragma unroll
        for (int u = 0; u < 4; ++u) {
            const int m = col0 + u * 16;
            float val = (acc[u][v] + bi) * sc + sh;
            if (relu) val = fmaxf(val, 0.0f);
            if (rrow) val += rrow[m / resDiv];
            orow[m] = val;
        }
    }
}

static inline void gemm(hipStream_t s,
    const float* W, int wRow, int wCol,
    const float* X, int ldx, int xB,
    float* Out, int ldo, int oB,
    const float* bias, int biasB,
    const float* scale, const float* shift, float scaleMul,
    const float* res, int ldr, int rB, int resDiv,
    int Cout, int Cin, int M, int relu, int batches)
{
    int tiles = (Cout / 16) * (M / 64);
    dim3 blk(256, 1, 1);                       // 8 waves per block
    dim3 grd((unsigned)((tiles + 7) / 8), (unsigned)batches, 1);
    wmma_gemm_kernel<<<grd, blk, 0, s>>>(W, wRow, wCol, X, ldx, xB, Out, ldo, oB,
                                         bias, biasB, scale, shift, scaleMul,
                                         res, ldr, rB, resDiv, Cout, Cin, M, relu);
}

// =====================================================================
// Small helper kernels
// =====================================================================

// weff[o,c] = w[o, c] + w[o, Cin + c]   (fold concat([x,x]) along K)
__global__ void fold_dual_kernel(const float* __restrict__ w, float* __restrict__ out,
                                 int Cout, int Cin)
{
    int i = blockIdx.x * blockDim.x + threadIdx.x;
    if (i >= Cout * Cin) return;
    int o = i / Cin, c = i - o * Cin;
    out[i] = w[o * (2 * Cin) + c] + w[o * (2 * Cin) + Cin + c];
}

// out[b,o] = bvec[o] + sum_f w[o, coff+f] * feat[b,f]   (fold concat([x,feat_rep]))
__global__ void feat_bias_kernel(const float* __restrict__ w, int ldw, int coff,
                                 const float* __restrict__ bvec,
                                 const float* __restrict__ feat, int featDim,
                                 float* __restrict__ out, int Cout)
{
    int b = blockIdx.x;
    int o = threadIdx.x;
    if (o >= Cout) return;
    float s = bvec[o];
    const float* fb = feat + b * featDim;
    const float* wr = w + o * ldw + coff;
    for (int f = 0; f < featDim; ++f) s += wr[f] * fb[f];
    out[b * Cout + o] = s;
}

// KNN: 16 nearest (incl. self) by d = |p|^2 + |q|^2 - 2 p.q ; ascending, stable
__global__ void knn_kernel(const float* __restrict__ xyz, int* __restrict__ idx)
{
    int g = blockIdx.x * blockDim.x + threadIdx.x;
    if (g >= Bb * Nn) return;
    int b = g / Nn, n = g - b * Nn;
    const float* base = xyz + b * 3 * Nn;
    float px = base[n], py = base[Nn + n], pz = base[2 * Nn + n];
    float sqp = px * px + py * py + pz * pz;
    float bd[Kk]; int bi[Kk];
#pragma unroll
    for (int j = 0; j < Kk; ++j) { bd[j] = 3.4e38f; bi[j] = 0; }
    for (int m = 0; m < Nn; ++m) {
        float qx = base[m], qy = base[Nn + m], qz = base[2 * Nn + m];
        float sqq = qx * qx + qy * qy + qz * qz;
        float d = sqp + sqq - 2.0f * (px * qx + py * qy + pz * qz);
        if (d < bd[Kk - 1]) {
            int j = Kk - 1;
            while (j > 0 && d < bd[j - 1]) { bd[j] = bd[j - 1]; bi[j] = bi[j - 1]; --j; }
            bd[j] = d; bi[j] = m;
        }
    }
#pragma unroll
    for (int j = 0; j < Kk; ++j) idx[g * Kk + j] = bi[j];
}

// pos_rel -> conv(3->64) -> BN -> relu ; h1[o, n*K+k]   (one batch)
__global__ void pos_stage1_kernel(const float* __restrict__ xyz, const int* __restrict__ idx,
                                  const float* __restrict__ w1, const float* __restrict__ b1,
                                  const float* __restrict__ g, const float* __restrict__ beta,
                                  float bnrs, float* __restrict__ h1, int b)
{
    int p = blockIdx.x * blockDim.x + threadIdx.x;   // 0 .. N*K
    if (p >= NK) return;
    int n = p >> 4, k = p & 15;
    const float* base = xyz + b * 3 * Nn;
    int m = idx[(b * Nn + n) * Kk + k];
    float dx = base[n] - base[m];
    float dy = base[Nn + n] - base[Nn + m];
    float dz = base[2 * Nn + n] - base[2 * Nn + m];
    for (int o = 0; o < DIMd; ++o) {
        float v = w1[o * 3 + 0] * dx + w1[o * 3 + 1] * dy + w1[o * 3 + 2] * dz + b1[o];
        v = v * (g[o] * bnrs) + beta[o];
        h1[o * NK + p] = fmaxf(v, 0.0f);
    }
}

// qk[c, n*K+k] = query[b,c,n] - key[b,c,idx] + pos_emb[c, n*K+k]   (one batch)
__global__ void qk_build_kernel(const float* __restrict__ query, const float* __restrict__ keyf,
                                const int* __restrict__ idx, const float* __restrict__ pe,
                                float* __restrict__ qk, int b)
{
    int t = blockIdx.x * blockDim.x + threadIdx.x;   // 0 .. 64*N*K
    if (t >= DIMd * NK) return;
    int c = t / NK, p = t - c * NK;
    int n = p >> 4, k = p & 15;
    int m = idx[(b * Nn + n) * Kk + k];
    qk[t] = query[(b * DIMd + c) * Nn + n]
          - keyf [(b * DIMd + c) * Nn + m]
          + pe[t];
    (void)k;
}

// agg[b,c,2n+r] = sum_k attn_r[c,n,k] * (value64[b,c,idx] + pos_emb[c,n,k]) (one batch)
__global__ void agg_kernel(const float* __restrict__ a0, const float* __restrict__ a1,
                           const float* __restrict__ val, const float* __restrict__ pe,
                           const int* __restrict__ idx, float* __restrict__ agg, int b)
{
    int t = blockIdx.x * blockDim.x + threadIdx.x;   // 0 .. 64*N
    if (t >= DIMd * Nn) return;
    int c = t / Nn, n = t - c * Nn;
    const int*   ib = idx + (b * Nn + n) * Kk;
    const float* vb = val + (b * DIMd + c) * Nn;
    const float* pp = pe + c * NK + n * Kk;
    const float* p0 = a0 + c * NK + n * Kk;
    const float* p1 = a1 + c * NK + n * Kk;
    float s0 = 0.0f, s1 = 0.0f;
#pragma unroll
    for (int k = 0; k < Kk; ++k) {
        float vg = vb[ib[k]] + pp[k];
        s0 += p0[k] * vg;
        s1 += p1[k] * vg;
    }
    float* ob = agg + (b * DIMd + c) * M2;
    ob[2 * n]     = s0;
    ob[2 * n + 1] = s1;
}

// completion = m4_w2(3x64) @ t + b2  -> d_out (B,3,2048)
__global__ void m4_final_kernel(const float* __restrict__ w2, const float* __restrict__ b2,
                                const float* __restrict__ t, float* __restrict__ out)
{
    int g = blockIdx.x * blockDim.x + threadIdx.x;   // 0 .. B*2048
    if (g >= Bb * M2) return;
    int b = g / M2, m = g - b * M2;
    const float* tb = t + b * DIMd * M2;
    for (int o = 0; o < 3; ++o) {
        float s = b2[o];
        for (int c = 0; c < DIMd; ++c) s += w2[o * DIMd + c] * tb[c * M2 + m];
        out[(b * 3 + o) * M2 + m] = s;
    }
}

// =====================================================================
// Host orchestration
// =====================================================================
extern "C" void kernel_launch(void* const* d_in, const int* in_sizes, int n_in,
                              void* d_out, int out_size, void* d_ws, size_t ws_size,
                              hipStream_t stream)
{
    (void)in_sizes; (void)n_in; (void)out_size; (void)ws_size;
    // ---- inputs (setup_inputs flattening order) ----
    const float* feat    = (const float*)d_in[0];
    const float* patch   = (const float*)d_in[1];
    const float* xyz     = (const float*)d_in[2];
    const float* v_w1    = (const float*)d_in[3];
    const float* v_b1    = (const float*)d_in[4];
    const float* v_w2    = (const float*)d_in[5];
    const float* v_b2    = (const float*)d_in[6];
    const float* v_ws    = (const float*)d_in[7];
    const float* v_bs    = (const float*)d_in[8];
    const float* q_w     = (const float*)d_in[9];
    const float* q_b     = (const float*)d_in[10];
    const float* k_w     = (const float*)d_in[11];
    const float* k_b     = (const float*)d_in[12];
    const float* vv_w    = (const float*)d_in[13];
    const float* vv_b    = (const float*)d_in[14];
    const float* pos_w1  = (const float*)d_in[15];
    const float* pos_b1  = (const float*)d_in[16];
    const float* pos_g   = (const float*)d_in[17];
    const float* pos_be  = (const float*)d_in[18];
    const float* pos_w2  = (const float*)d_in[19];
    const float* pos_b2  = (const float*)d_in[20];
    const float* at_w1   = (const float*)d_in[21];
    const float* at_b1   = (const float*)d_in[22];
    const float* at_g    = (const float*)d_in[23];
    const float* at_be   = (const float*)d_in[24];
    const float* at_wt   = (const float*)d_in[25];   // (HID, DIM, UP)
    const float* at_bt   = (const float*)d_in[26];
    const float* end_w   = (const float*)d_in[27];
    const float* end_b   = (const float*)d_in[28];
    const float* m1_w1   = (const float*)d_in[29];
    const float* m1_b1   = (const float*)d_in[30];
    const float* m1_w2   = (const float*)d_in[31];
    const float* m1_b2   = (const float*)d_in[32];
    const float* m1_ws   = (const float*)d_in[33];
    const float* m1_bs   = (const float*)d_in[34];
    const float* m2_w1   = (const float*)d_in[35];
    const float* m2_b1   = (const float*)d_in[36];
    const float* m2_w2   = (const float*)d_in[37];
    const float* m2_b2   = (const float*)d_in[38];
    const float* m2_ws   = (const float*)d_in[39];
    const float* m2_bs   = (const float*)d_in[40];
    const float* m3_w1   = (const float*)d_in[41];
    const float* m3_b1   = (const float*)d_in[42];
    const float* m3_w2   = (const float*)d_in[43];
    const float* m3_b2   = (const float*)d_in[44];
    const float* m3_ws   = (const float*)d_in[45];
    const float* m3_bs   = (const float*)d_in[46];
    const float* m4_w1   = (const float*)d_in[47];
    const float* m4_b1   = (const float*)d_in[48];
    const float* m4_w2   = (const float*)d_in[49];
    const float* m4_b2   = (const float*)d_in[50];

    // ---- workspace layout (floats; idx ints first) ----
    int*   idx = (int*)d_ws;
    float* wsf = (float*)d_ws;
    const long long OFF_VAL128 = (long long)Bb * Nn * Kk;                  // after idx ints
    const long long OFF_KEYF   = OFF_VAL128 + (long long)Bb * CINc * Nn;
    const long long OFF_QUERY  = OFF_KEYF   + (long long)Bb * DIMd * Nn;
    const long long OFF_VAL64  = OFF_QUERY  + (long long)Bb * DIMd * Nn;
    const long long OFF_WEFF1  = OFF_VAL64  + (long long)Bb * DIMd * Nn;
    const long long OFF_WEFFS  = OFF_WEFF1  + CINc * CINc;
    const long long OFF_B1M1   = OFF_WEFFS  + CINc * CINc;
    const long long OFF_BSM1   = OFF_B1M1   + Bb * CINc;
    const long long OFF_B1M3   = OFF_BSM1   + Bb * CINc;
    const long long OFF_BSM3   = OFF_B1M3   + Bb * CINc;
    const long long OFF_H1     = OFF_BSM3   + Bb * CINc;                   // per-batch, reused
    const long long OFF_PE     = OFF_H1     + (long long)DIMd * NK;
    const long long OFF_QK     = OFF_PE     + (long long)DIMd * NK;
    const long long OFF_H      = OFF_QK     + (long long)DIMd * NK;
    const long long OFF_A0     = OFF_H      + (long long)HIDh * NK;
    const long long OFF_A1     = OFF_A0     + (long long)DIMd * NK;
    const long long OFF_AGG    = OFF_A1     + (long long)DIMd * NK;
    const long long OFF_X1     = OFF_AGG    + (long long)Bb * DIMd * M2;
    const long long OFF_T      = OFF_X1     + (long long)Bb * CINc * M2;
    const long long OFF_R1     = OFF_T      + (long long)Bb * CINc * M2;
    const long long OFF_R2     = OFF_R1     + (long long)Bb * CINc * M2;

    float* VAL128 = wsf + OFF_VAL128;
    float* KEYF   = wsf + OFF_KEYF;
    float* QUERY  = wsf + OFF_QUERY;
    float* VAL64  = wsf + OFF_VAL64;
    float* WEFF1  = wsf + OFF_WEFF1;
    float* WEFFS  = wsf + OFF_WEFFS;
    float* B1M1   = wsf + OFF_B1M1;
    float* BSM1   = wsf + OFF_BSM1;
    float* B1M3   = wsf + OFF_B1M3;
    float* BSM3   = wsf + OFF_BSM3;
    float* H1     = wsf + OFF_H1;
    float* PE     = wsf + OFF_PE;
    float* QK     = wsf + OFF_QK;
    float* Hbuf   = wsf + OFF_H;
    float* A0     = wsf + OFF_A0;
    float* A1     = wsf + OFF_A1;
    float* AGG    = wsf + OFF_AGG;
    float* X1     = wsf + OFF_X1;
    float* Tbuf   = wsf + OFF_T;
    float* R1     = wsf + OFF_R1;
    float* R2     = wsf + OFF_R2;

    float* outC  = (float*)d_out;                            // completion (B,3,2048)
    float* outX3 = (float*)d_out + (long long)Bb * 3 * M2;   // x3 (B,128,2048)

    const float bnrs = 1.0f / sqrtf(1.0f + 1e-5f);
    const int pB = CINc * Nn;                // patch batch stride
    const int dB = DIMd * Nn;

    // ---- 0) weight/bias folding + knn ----
    fold_dual_kernel<<<(CINc * CINc + 255) / 256, 256, 0, stream>>>(v_w1, WEFF1, CINc, CINc);
    fold_dual_kernel<<<(CINc * CINc + 255) / 256, 256, 0, stream>>>(v_ws, WEFFS, CINc, CINc);
    feat_bias_kernel<<<Bb, CINc, 0, stream>>>(m1_w1, CINc + FEATf, CINc, m1_b1, feat, FEATf, B1M1, CINc);
    feat_bias_kernel<<<Bb, CINc, 0, stream>>>(m1_ws, CINc + FEATf, CINc, m1_bs, feat, FEATf, BSM1, CINc);
    feat_bias_kernel<<<Bb, CINc, 0, stream>>>(m3_w1, CINc + FEATf, CINc, m3_b1, feat, FEATf, B1M3, CINc);
    feat_bias_kernel<<<Bb, CINc, 0, stream>>>(m3_ws, CINc + FEATf, CINc, m3_bs, feat, FEATf, BSM3, CINc);
    knn_kernel<<<(Bb * Nn + 127) / 128, 128, 0, stream>>>(xyz, idx);

    // ---- 1) mlp_v (folded): value128 = w2@relu(weff1@patch+b1) + weffs@patch+bs ----
    gemm(stream, WEFF1, CINc, 1, patch, Nn, pB, Tbuf, Nn, pB,
         v_b1, 0, nullptr, nullptr, 1.0f, nullptr, 0, 0, 1, CINc, CINc, Nn, 1, Bb);
    gemm(stream, WEFFS, CINc, 1, patch, Nn, pB, VAL128, Nn, pB,
         v_bs, 0, nullptr, nullptr, 1.0f, nullptr, 0, 0, 1, CINc, CINc, Nn, 0, Bb);
    gemm(stream, v_w2, CINc, 1, Tbuf, Nn, pB, VAL128, Nn, pB,
         v_b2, 0, nullptr, nullptr, 1.0f, VAL128, Nn, pB, 1, CINc, CINc, Nn, 0, Bb);

    // ---- 2) q/k/v projections ----
    gemm(stream, k_w,  CINc, 1, patch,  Nn, pB, KEYF,  Nn, dB,
         k_b, 0, nullptr, nullptr, 1.0f, nullptr, 0, 0, 1, DIMd, CINc, Nn, 0, Bb);
    gemm(stream, q_w,  CINc, 1, patch,  Nn, pB, QUERY, Nn, dB,
         q_b, 0, nullptr, nullptr, 1.0f, nullptr, 0, 0, 1, DIMd, CINc, Nn, 0, Bb);
    gemm(stream, vv_w, CINc, 1, VAL128, Nn, pB, VAL64, Nn, dB,
         vv_b, 0, nullptr, nullptr, 1.0f, nullptr, 0, 0, 1, DIMd, CINc, Nn, 0, Bb);

    // ---- 3) attention (per-batch loop, buffers reused) ----
    for (int b = 0; b < Bb; ++b) {
        pos_stage1_kernel<<<(NK + 255) / 256, 256, 0, stream>>>(
            xyz, idx, pos_w1, pos_b1, pos_g, pos_be, bnrs, H1, b);
        gemm(stream, pos_w2, DIMd, 1, H1, NK, 0, PE, NK, 0,
             pos_b2, 0, nullptr, nullptr, 1.0f, nullptr, 0, 0, 1, DIMd, DIMd, NK, 0, 1);
        qk_build_kernel<<<(DIMd * NK + 255) / 256, 256, 0, stream>>>(
            QUERY, KEYF, idx, PE, QK, b);
        // h = relu(BN(attn_w1 @ qk + b1))
        gemm(stream, at_w1, DIMd, 1, QK, NK, 0, Hbuf, NK, 0,
             at_b1, 0, at_g, at_be, bnrs, nullptr, 0, 0, 1, HIDh, DIMd, NK, 1, 1);
        // transpose conv: attn_r = wt[:, :, r]^T @ h + bt
        gemm(stream, at_wt + 0, 2, 2 * DIMd, Hbuf, NK, 0, A0, NK, 0,
             at_bt, 0, nullptr, nullptr, 1.0f, nullptr, 0, 0, 1, DIMd, HIDh, NK, 0, 1);
        gemm(stream, at_wt + 1, 2, 2 * DIMd, Hbuf, NK, 0, A1, NK, 0,
             at_bt, 0, nullptr, nullptr, 1.0f, nullptr, 0, 0, 1, DIMd, HIDh, NK, 0, 1);
        agg_kernel<<<(DIMd * Nn + 255) / 256, 256, 0, stream>>>(
            A0, A1, VAL64, PE, idx, AGG, b);
    }

    // ---- 4) end conv + residual repeat(identity, UP) ----
    gemm(stream, end_w, DIMd, 1, AGG, M2, DIMd * M2, X1, M2, CINc * M2,
         end_b, 0, nullptr, nullptr, 1.0f, VAL128, Nn, pB, 2, CINc, DIMd, M2, 0, Bb);

    const int cB2 = CINc * M2;
    const int dB2 = DIMd * M2;

    // ---- 5) mlp1 (feat concat folded into per-batch bias) ----
    gemm(stream, m1_w1, CINc + FEATf, 1, X1, M2, cB2, Tbuf, M2, cB2,
         B1M1, CINc, nullptr, nullptr, 1.0f, nullptr, 0, 0, 1, CINc, CINc, M2, 1, Bb);
    gemm(stream, m1_ws, CINc + FEATf, 1, X1, M2, cB2, R1, M2, cB2,
         BSM1, CINc, nullptr, nullptr, 1.0f, nullptr, 0, 0, 1, CINc, CINc, M2, 0, Bb);
    gemm(stream, m1_w2, CINc, 1, Tbuf, M2, cB2, R1, M2, cB2,
         m1_b2, 0, nullptr, nullptr, 1.0f, R1, M2, cB2, 1, CINc, CINc, M2, 0, Bb);

    // ---- 6) mlp2 ----
    gemm(stream, m2_w1, CINc, 1, R1, M2, cB2, Tbuf, M2, dB2,
         m2_b1, 0, nullptr, nullptr, 1.0f, nullptr, 0, 0, 1, DIMd, CINc, M2, 1, Bb);
    gemm(stream, m2_ws, CINc, 1, R1, M2, cB2, R2, M2, cB2,
         m2_bs, 0, nullptr, nullptr, 1.0f, nullptr, 0, 0, 1, CINc, CINc, M2, 0, Bb);
    gemm(stream, m2_w2, DIMd, 1, Tbuf, M2, dB2, R2, M2, cB2,
         m2_b2, 0, nullptr, nullptr, 1.0f, R2, M2, cB2, 1, CINc, DIMd, M2, 0, Bb);

    // ---- 7) mlp3 -> x3 (second output) ----
    gemm(stream, m3_w1, CINc + FEATf, 1, R2, M2, cB2, Tbuf, M2, cB2,
         B1M3, CINc, nullptr, nullptr, 1.0f, nullptr, 0, 0, 1, CINc, CINc, M2, 1, Bb);
    gemm(stream, m3_ws, CINc + FEATf, 1, R2, M2, cB2, outX3, M2, cB2,
         BSM3, CINc, nullptr, nullptr, 1.0f, nullptr, 0, 0, 1, CINc, CINc, M2, 0, Bb);
    gemm(stream, m3_w2, CINc, 1, Tbuf, M2, cB2, outX3, M2, cB2,
         m3_b2, 0, nullptr, nullptr, 1.0f, outX3, M2, cB2, 1, CINc, CINc, M2, 0, Bb);

    // ---- 8) completion head ----
    gemm(stream, m4_w1, CINc, 1, outX3, M2, cB2, Tbuf, M2, dB2,
         m4_b1, 0, nullptr, nullptr, 1.0f, nullptr, 0, 0, 1, DIMd, CINc, M2, 1, Bb);
    m4_final_kernel<<<(Bb * M2 + 255) / 256, 256, 0, stream>>>(m4_w2, m4_b2, Tbuf, outC);
}
